// ChannelCrossAttention_81681688036024
// MI455X (gfx1250) — matrix operand order repaired and checked by reading, not compiled, not run on hardware
//
#include <hip/hip_runtime.h>
#include <hip/hip_bf16.h>

// ---------------------------------------------------------------------------
// ChannelCrossAttention for MI455X (gfx1250, wave32, WMMA).
//
// Pipeline (all on `stream`):
//   1) proj_kernel<TRANS=1>: Qh[b][n][32]  = Wq·feat1 + bq   (bf16, d-contig)
//   2) proj_kernel<TRANS=1>: Kh[b][n][32]  = Wk·feat2 + bk   (bf16, d-contig)
//   3) proj_kernel<TRANS=0>: Vt[b][c][N]   = Wv·feat2 + bv   (bf16, key-contig)
//   4) attn_kernel: flash-attention (online softmax) with
//      v_wmma_f32_16x16x32_bf16 for both Q·K^T and P·V; fused epilogue
//      out = gamma * (O / l) + feat1.
//
// Never materializes the 268 MB energy tensor.
// ---------------------------------------------------------------------------

typedef __attribute__((ext_vector_type(16))) __bf16 v16bf;
typedef __attribute__((ext_vector_type(8)))  __bf16 v8bf;
typedef __attribute__((ext_vector_type(8)))  float  v8f;

#define CCA_B 4
#define CCA_C 256
#define CCA_N 4096   // H*W
#define CCA_D 32     // C/8

__device__ __forceinline__ float fast_exp2(float x) {
#if __has_builtin(__builtin_amdgcn_exp2f)
  return __builtin_amdgcn_exp2f(x);
#else
  return exp2f(x);
#endif
}

// ---------------------------------------------------------------------------
// 1x1-conv projection: y = W x + b, x:[B][C][N] f32 row-per-channel.
// TRANS=1 -> y[b][n][O] (d-contiguous rows for WMMA operands)
// TRANS=0 -> y[b][O][N] (key-contiguous rows, natural V layout)
// ---------------------------------------------------------------------------
template <bool TRANS>
__global__ __launch_bounds__(256) void proj_kernel(
    const float* __restrict__ x, const float* __restrict__ W,
    const float* __restrict__ bias, __bf16* __restrict__ y, int O)
{
  const int t = blockIdx.x * 256 + threadIdx.x;   // n fastest -> coalesced x
  const int n = t & (CCA_N - 1);
  const int o = (t >> 12) % O;
  const int b = t / (CCA_N * O);

  float a = bias[o];
  const float* xb = x + (size_t)b * CCA_C * CCA_N + n;
  const float* w  = W + (size_t)o * CCA_C;
#pragma unroll 8
  for (int c = 0; c < CCA_C; ++c)
    a = fmaf(w[c], xb[(size_t)c * CCA_N], a);

  if (TRANS) y[((size_t)b * CCA_N + n) * O + o] = (__bf16)a;
  else       y[((size_t)b * O + o) * CCA_N + n] = (__bf16)a;
}

// ---------------------------------------------------------------------------
// Flash attention. Grid: B * (N/64) blocks, 256 threads = 8 waves.
// Wave w: row-tile rt = w>>1 (16 query rows), col-half ch = w&1 (128 channels).
// Per wave: 8 accumulator tiles (16x16 f32), running max/sum per row in the
// C/D broadcast layout (VGPR v <-> rows v / v+8 per lane-half).
// ---------------------------------------------------------------------------
__global__ __launch_bounds__(256) void attn_kernel(
    const __bf16* __restrict__ Qh,   // [B][N][32]
    const __bf16* __restrict__ Kh,   // [B][N][32]
    const __bf16* __restrict__ Vt,   // [B][C][N]
    const float*  __restrict__ feat1,// [B][C][N]
    const float*  __restrict__ gamma_p,
    float* __restrict__ out)         // [B][C][N]
{
  constexpr float LOG2E = 1.4426950408889634f;
  __shared__ __align__(16) __bf16 Pbuf[4][16][64];   // 8 KB P staging

  const int b    = blockIdx.x >> 6;
  const int q0   = (blockIdx.x & 63) * 64;
  const int lane = threadIdx.x & 31;
  const int wave = threadIdx.x >> 5;
  const int rt   = wave >> 1;
  const int ch   = wave & 1;
  const int half = lane >> 4;      // which 16-lane half
  const int l16  = lane & 15;

  // ---- Q A-operand (16x32 bf16): lane = row l16; K-chunks per ISA layout ----
  const int qrow = q0 + rt * 16 + l16;
  const __bf16* qp = Qh + ((size_t)b * CCA_N + qrow) * CCA_D;
  union { v16bf v; v8bf h[2]; } qa;
  qa.h[0] = *(const v8bf*)(qp + half * 8);        // K = 0..7  / 8..15
  qa.h[1] = *(const v8bf*)(qp + 16 + half * 8);   // K = 16..23 / 24..31

  v8f  acc[8];
  float m_i[8], l_i[8];
  {
    v8f zf = {};
#pragma unroll
    for (int i = 0; i < 8; ++i) { acc[i] = zf; m_i[i] = -1e30f; l_i[i] = 0.f; }
  }

  const __bf16* Kb = Kh + (size_t)b * CCA_N * CCA_D;
  const __bf16* Vb = Vt + (size_t)b * CCA_C * CCA_N;

  for (int kt = 0; kt < CCA_N; kt += 64) {
    // ---- S = Q K^T (16 rows x 64 keys): 4 WMMAs, K-dim = d = 32 ----
    v8f s[4];
#pragma unroll
    for (int t = 0; t < 4; ++t) {
      // B-operand 32x16: lane = column (key), d = 8*half..+7 and 16+8*half..+7
      const __bf16* kp = Kb + (size_t)(kt + 16 * t + l16) * CCA_D + half * 8;
      union { v16bf v; v8bf h[2]; } kb;
      kb.h[0] = *(const v8bf*)(kp);
      kb.h[1] = *(const v8bf*)(kp + 16);
      v8f z = {};
      s[t] = __builtin_amdgcn_wmma_f32_16x16x32_bf16(
                 false, qa.v, false, kb.v, (short)0, z, false, false);
    }

    // ---- online softmax (per C/D VGPR index v = rows v / v+8) ----
    float scale[8];
#pragma unroll
    for (int v = 0; v < 8; ++v) {
      float x = fmaxf(fmaxf(s[0][v], s[1][v]), fmaxf(s[2][v], s[3][v]));
      x = fmaxf(x, __shfl_xor(x, 1));
      x = fmaxf(x, __shfl_xor(x, 2));
      x = fmaxf(x, __shfl_xor(x, 4));
      x = fmaxf(x, __shfl_xor(x, 8));   // row-max, stays within 16-lane half
      const float mn = fmaxf(m_i[v], x);
      scale[v] = fast_exp2((m_i[v] - mn) * LOG2E);
      m_i[v] = mn;
      float ps = 0.f;
#pragma unroll
      for (int t = 0; t < 4; ++t) {
        const float p = fast_exp2((s[t][v] - mn) * LOG2E);
        s[t][v] = p;
        ps += p;
      }
      ps += __shfl_xor(ps, 1);
      ps += __shfl_xor(ps, 2);
      ps += __shfl_xor(ps, 4);
      ps += __shfl_xor(ps, 8);
      l_i[v] = l_i[v] * scale[v] + ps;
#pragma unroll
      for (int ct = 0; ct < 8; ++ct) acc[ct][v] *= scale[v];
    }

    // ---- stage P (bf16, row-major 16x64) through LDS; dedupe wave pairs ----
    __syncthreads();                 // previous iteration's P reads complete
    if (ch == 0) {
#pragma unroll
      for (int t = 0; t < 4; ++t)
#pragma unroll
        for (int v = 0; v < 8; ++v)
          Pbuf[rt][v + 8 * half][16 * t + l16] = (__bf16)s[t][v];
    }
    __syncthreads();

    // ---- reload P in A-operand layout (keys 0..31 and 32..63) ----
    const __bf16* pp = &Pbuf[rt][l16][0];
    union { v16bf v; v8bf h[2]; } pa0, pa1;
    pa0.h[0] = *(const v8bf*)(pp + half * 8);
    pa0.h[1] = *(const v8bf*)(pp + 16 + half * 8);
    pa1.h[0] = *(const v8bf*)(pp + 32 + half * 8);
    pa1.h[1] = *(const v8bf*)(pp + 48 + half * 8);

    // ---- O += P V : 8 col tiles x 2 WMMAs (K = 32 keys each) ----
#pragma unroll
    for (int ct = 0; ct < 8; ++ct) {
      const int c = 128 * ch + 16 * ct + l16;      // B-operand column
      const __bf16* vp = Vb + (size_t)c * CCA_N + kt + half * 8;
      union { v16bf v; v8bf h[2]; } vb0, vb1;
      vb0.h[0] = *(const v8bf*)(vp);
      vb0.h[1] = *(const v8bf*)(vp + 16);
      vb1.h[0] = *(const v8bf*)(vp + 32);
      vb1.h[1] = *(const v8bf*)(vp + 48);
      acc[ct] = __builtin_amdgcn_wmma_f32_16x16x32_bf16(
                    false, pa0.v, false, vb0.v, (short)0, acc[ct], false, false);
      acc[ct] = __builtin_amdgcn_wmma_f32_16x16x32_bf16(
                    false, pa1.v, false, vb1.v, (short)0, acc[ct], false, false);
    }
  }

  // ---- epilogue: out = gamma * (acc / l) + feat1 ----
  const float g = gamma_p[0];
#pragma unroll
  for (int v = 0; v < 8; ++v) {
    const float inv = 1.0f / l_i[v];
    const int m = q0 + rt * 16 + v + 8 * half;
#pragma unroll
    for (int ct = 0; ct < 8; ++ct) {
      const int c = 128 * ch + 16 * ct + l16;
      const size_t idx = ((size_t)b * CCA_C + c) * CCA_N + m;
      out[idx] = g * (acc[ct][v] * inv) + feat1[idx];
    }
  }
}

// ---------------------------------------------------------------------------
extern "C" void kernel_launch(void* const* d_in, const int* in_sizes, int n_in,
                              void* d_out, int out_size, void* d_ws, size_t ws_size,
                              hipStream_t stream) {
  (void)in_sizes; (void)n_in; (void)out_size; (void)ws_size;
  const float* feat1 = (const float*)d_in[0];
  const float* feat2 = (const float*)d_in[1];
  const float* Wq    = (const float*)d_in[2];
  const float* bq    = (const float*)d_in[3];
  const float* Wk    = (const float*)d_in[4];
  const float* bk    = (const float*)d_in[5];
  const float* Wv    = (const float*)d_in[6];
  const float* bv    = (const float*)d_in[7];
  const float* gamma = (const float*)d_in[8];
  float* out = (float*)d_out;

  // Workspace: Qh (1 MB) | Kh (1 MB) | Vt (8 MB), all bf16.
  __bf16* Qh = (__bf16*)d_ws;
  __bf16* Kh = Qh + (size_t)CCA_B * CCA_N * CCA_D;
  __bf16* Vt = Kh + (size_t)CCA_B * CCA_N * CCA_D;

  const int qk_threads = CCA_B * CCA_D * CCA_N;   // 524288
  const int v_threads  = CCA_B * CCA_C * CCA_N;   // 4194304
  proj_kernel<true ><<<qk_threads / 256, 256, 0, stream>>>(feat1, Wq, bq, Qh, CCA_D);
  proj_kernel<true ><<<qk_threads / 256, 256, 0, stream>>>(feat2, Wk, bk, Kh, CCA_D);
  proj_kernel<false><<<v_threads  / 256, 256, 0, stream>>>(feat2, Wv, bv, Vt, CCA_C);

  attn_kernel<<<CCA_B * (CCA_N / 64), 256, 0, stream>>>(Qh, Kh, Vt, feat1, gamma, out);
}
